// PygAtt_55516747268136
// MI455X (gfx1250) — matrix operand order repaired and checked by reading, not compiled, not run on hardware
//
#include <hip/hip_runtime.h>

typedef __attribute__((ext_vector_type(16))) _Float16 v16h;
typedef __attribute__((ext_vector_type(8)))  float    v8f;

#define HEADS 4
#define F 128               // OUT_FEATURES
#define NEG_SLOPE 0.02f

__device__ __forceinline__ float leaky(float a) {
  return a > 0.f ? a : NEG_SLOPE * a;
}

// Sign-split float atomic max: correct total order across mixed signs when
// the cell is initialized to -inf. Emits real global atomics (max_i32/min_u32).
__device__ __forceinline__ void atomicMaxF(float* addr, float val) {
  if (val >= 0.f)
    atomicMax((int*)addr, __float_as_int(val));
  else
    atomicMin((unsigned int*)addr, __float_as_uint(val));
}

// ---------------- pass A: init out / amax / denom ----------------
__global__ void k_init(float* __restrict__ out, float* __restrict__ amax,
                       float* __restrict__ denom, int N) {
  int idx = blockIdx.x * blockDim.x + threadIdx.x;
  if (idx < N * F) out[idx] = 0.f;
  if (idx < N * HEADS) {
    amax[idx]  = -__builtin_inff();
    denom[idx] = 0.f;
  }
}

// ---------------- pass B: S[N,8] = X @ Wmat via v_wmma_f32_16x16x32_f16 ----
// Wmat[128,8]: col h   holds W[0:32]  at rows h*32..h*32+31  -> s[n,h]
//              col 4+h holds W[32:64] at rows h*32..h*32+31  -> t[n,h]
// One 16-node tile per wave; K-loop 128/32 = 4 WMMAs, f32 accumulate.
//
// B-operand insight: for lane column n = lane&15, the column's nonzero 32-row
// block lives entirely in K-iteration q = n&3 (n<8; columns 8..15 are zero
// padding), and this lane's 16 B-halves of it are the floats
// W[b+8*(lane>>4) .. +8) and W[b+16+8*(lane>>4) .. +8), b = n<4?0:32 -- loop-invariant. So build one masked
// per-lane vector up front and select it in (or zero) per iteration.
__global__ void k_scores_wmma(const float* __restrict__ x,
                              const float* __restrict__ W,
                              float* __restrict__ S, int nTiles) {
  const int t    = threadIdx.x;
  const int wave = t >> 5;
  const int lane = t & 31;
  const int tile = blockIdx.x * (blockDim.x >> 5) + wave;
  if (tile >= nTiles) return;                  // wave-uniform; EXEC stays full

  const int m  = lane & 15;                    // A row (M) / B,D column (N)
  const int hi = lane >> 4;                    // lane half selector

  // ---- per-lane constant B column slice (branchless) ----
  const int   woff = ((m >= 4 && m < 8) ? 32 : 0) + hi * 8;  // in [0,40]
  const float msk  = (m < 8) ? 1.f : 0.f;
  float wtmp[16];
  {
    const float4* wp = (const float4*)(W + woff);
    *(float4*)(wtmp + 0)  = wp[0];
    *(float4*)(wtmp + 4)  = wp[1];
    *(float4*)(wtmp + 8)  = wp[4];
    *(float4*)(wtmp + 12) = wp[5];
  }
  v16h vb;
#pragma unroll
  for (int p = 0; p < 16; ++p) vb[p] = (_Float16)(msk * wtmp[p]);
  const int  qsel   = (m < 8) ? (m & 3) : -1;  // which K-iteration uses vb
  const v16h zero16 = {};

  const float* xrow = x + (size_t)(tile * 16 + m) * F;
  v8f acc = {};
#pragma unroll
  for (int q = 0; q < 4; ++q) {
    const int kBase = q * 32;
    // ---- A tile: 16x32 f16. Lane<16: K in [0,8)+[16,24); lane>=16: +8. ----
    float atmp[16];
    {
      const float4* xp = (const float4*)(xrow + kBase + hi * 8);
      const float4* xq = (const float4*)(xrow + kBase + 16 + hi * 8);
      *(float4*)(atmp + 0)  = xp[0];
      *(float4*)(atmp + 4)  = xp[1];
      *(float4*)(atmp + 8)  = xq[0];
      *(float4*)(atmp + 12) = xq[1];
    }
    v16h a;
#pragma unroll
    for (int p = 0; p < 16; ++p) a[p] = (_Float16)atmp[p];

    const v16h b = (q == qsel) ? vb : zero16;  // single vector cndmask
    acc = __builtin_amdgcn_wmma_f32_16x16x32_f16(false, a, false, b,
                                                 (short)0, acc, false, false);
  }

  // D layout: VGPR r -> M = r + 8*hi, N = lane%16. Only cols 0..7 are real.
  if (m < 8) {
    const int mBase = tile * 16 + hi * 8;
#pragma unroll
    for (int r = 0; r < 8; ++r)
      S[(size_t)(mBase + r) * 8 + m] = acc[r];
  }
}

// ---------------- pass C: segment max over targets ----------------
__global__ void k_segmax(const float* __restrict__ S,
                         const long long* __restrict__ ei,
                         float* __restrict__ amax, int E) {
  int idx = blockIdx.x * blockDim.x + threadIdx.x;
  if (idx >= E * HEADS) return;
  const int e = idx >> 2, h = idx & 3;
  const int i = (int)ei[e];
  const int j = (int)ei[E + e];
  const float a = leaky(S[(size_t)i * 8 + h] + S[(size_t)j * 8 + 4 + h]);
  atomicMaxF(&amax[i * HEADS + h], a);
}

// ---------------- pass D: segment sum of exp(alpha - amax) ----------------
__global__ void k_segsum(const float* __restrict__ S,
                         const long long* __restrict__ ei,
                         const float* __restrict__ amax,
                         float* __restrict__ denom, int E) {
  int idx = blockIdx.x * blockDim.x + threadIdx.x;
  if (idx >= E * HEADS) return;
  const int e = idx >> 2, h = idx & 3;
  const int i = (int)ei[e];
  const int j = (int)ei[E + e];
  const float a = leaky(S[(size_t)i * 8 + h] + S[(size_t)j * 8 + 4 + h]);
  const float w = expf(a - amax[i * HEADS + h]);
  atomicAdd(&denom[i * HEADS + h], w);
}

// ---------------- pass E: weighted scatter aggregation --------------------
// One wave per edge. Lane l owns channels [4l,4l+4) -> head = l>>3.
// float4 gather of x[j] (L2-resident), 4 non-returning f32 atomic adds.
__global__ void k_aggregate(const float* __restrict__ x,
                            const float* __restrict__ S,
                            const long long* __restrict__ ei,
                            const float* __restrict__ amax,
                            const float* __restrict__ denom,
                            float* __restrict__ out, int E) {
  const int wave = threadIdx.x >> 5;
  const int lane = threadIdx.x & 31;
  const int e = blockIdx.x * (blockDim.x >> 5) + wave;
  if (e >= E) return;
  const int i = (int)ei[e];
  const int j = (int)ei[E + e];
  const int h = lane >> 3;
  const float a    = leaky(S[(size_t)i * 8 + h] + S[(size_t)j * 8 + 4 + h]);
  const float coef = expf(a - amax[i * HEADS + h]) /
                     (denom[i * HEADS + h] + 1e-16f);
  const float4 xv = ((const float4*)(x + (size_t)j * F))[lane];
  float* dst = out + (size_t)i * F + lane * 4;
  atomicAdd(dst + 0, xv.x * coef);
  atomicAdd(dst + 1, xv.y * coef);
  atomicAdd(dst + 2, xv.z * coef);
  atomicAdd(dst + 3, xv.w * coef);
}

extern "C" void kernel_launch(void* const* d_in, const int* in_sizes, int n_in,
                              void* d_out, int out_size, void* d_ws, size_t ws_size,
                              hipStream_t stream) {
  const float*     x  = (const float*)d_in[0];
  const long long* ei = (const long long*)d_in[1];   // int64 [2,E]
  const float*     W  = (const float*)d_in[2];       // 64 floats
  float* out = (float*)d_out;

  const int N = in_sizes[0] / F;     // 50000
  const int E = in_sizes[1] / 2;     // 800000

  // Scratch layout: S[N*8] | amax[N*4] | denom[N*4]  (~3.2 MB total)
  float* S     = (float*)d_ws;
  float* amax  = S + (size_t)N * 8;
  float* denom = amax + (size_t)N * HEADS;

  {
    const int total = N * F;
    k_init<<<(total + 255) / 256, 256, 0, stream>>>(out, amax, denom, N);
  }
  {
    const int nTiles = (N + 15) / 16;              // 3125, exact
    const int wpb = 4;                             // waves per block
    k_scores_wmma<<<(nTiles + wpb - 1) / wpb, wpb * 32, 0, stream>>>(x, W, S, nTiles);
  }
  {
    const int total = E * HEADS;
    k_segmax<<<(total + 255) / 256, 256, 0, stream>>>(S, ei, amax, E);
    k_segsum<<<(total + 255) / 256, 256, 0, stream>>>(S, ei, amax, denom, E);
  }
  {
    const int wpb = 8;                             // 8 edges (waves) per block
    k_aggregate<<<(E + wpb - 1) / wpb, wpb * 32, 0, stream>>>(x, S, ei, amax,
                                                              denom, out, E);
  }
}
